// GAT_Top_34230889349733
// MI455X (gfx1250) — matrix operand order repaired and checked by reading, not compiled, not run on hardware
//
#include <hip/hip_runtime.h>
#include <hip/hip_bf16.h>
#include <float.h>

// ---------------------------------------------------------------------------
// MI455X (gfx1250) GAT pipeline.
// GEMMs: v_wmma_f32_16x16x32_bf16, 4 accumulators (16x64 tile) per wave,
// activations pre-converted to bf16 once (halves A-side HBM traffic).
// All 4 B fragments are loaded before the 4 wmmas so the scheduler can use
// partial s_wait_loadcnt and overlap loads with the matrix pipe.
// Graph gather/scatter: fp32 + atomics (bandwidth bound, ~0.5-1 GB traffic).
// GAT bias is dropped: per-column constant shift cancels exactly in batchnorm.
// ---------------------------------------------------------------------------

typedef __attribute__((ext_vector_type(16))) __bf16         v16bf;
typedef __attribute__((ext_vector_type(8)))  float          v8f;
typedef __attribute__((ext_vector_type(16))) unsigned short v16us;
typedef __attribute__((ext_vector_type(8)))  unsigned short v8us;

__device__ __forceinline__ unsigned short f2bf(float f) {
  unsigned int u = __float_as_uint(f);
  unsigned int r = u + 0x7FFFu + ((u >> 16) & 1u);   // round-to-nearest-even
  return (unsigned short)(r >> 16);
}

__device__ __forceinline__ void atomicMaxF(float* addr, float v) {
  if (v >= 0.0f) atomicMax((int*)addr, __float_as_int(v));
  else           atomicMin((unsigned int*)addr, __float_as_uint(v));
}

// ---------------------------------------------------------------------------
// WMMA GEMM: C[M x Nc] = A[M x K] (bf16 row-major) * B (bf16, pre-transposed
// to Nc x K) + bias. Each wave computes a 16x64 tile: one A fragment feeds
// 4 accumulators (4 v_wmma per k-step). 2 waves/block -> 16x128 per block.
// A frag: lanes 0-15 row M=l16 hold K = k0+0..7 (elems 0-7), k0+16..23 (8-15);
//         lanes 16-31 hold K = k0+8..15 / k0+24..31.  (ISA 16-bit A 16x32)
// B frag: lane column N=l16; same K order as A: lanes 0-15 hold K = k0+0..7,
//         k0+16..23; lanes 16-31 hold K = k0+8..15 / k0+24..31.
// C frag: elem r -> C[16*tm + 8*half + r][16*tn + l16]
// ---------------------------------------------------------------------------
__global__ void __launch_bounds__(64)
gemm_wmma_bf16(const unsigned short* __restrict__ A,
               const unsigned short* __restrict__ Bt,
               const float* __restrict__ bias, float* __restrict__ C,
               int M, int K, int Ncols)
{
  const int lane = threadIdx.x & 31;
  const int wv   = threadIdx.x >> 5;
  const int hf   = lane >> 4;
  const int l16  = lane & 15;
  const int tn0  = (blockIdx.x * 2 + wv) * 4;   // first of 4 consecutive n-tiles
  const int tm   = blockIdx.y;
  const int m    = tm * 16 + l16;

  v8f acc[4];
#pragma unroll
  for (int t = 0; t < 4; ++t) acc[t] = (v8f){0.f,0.f,0.f,0.f,0.f,0.f,0.f,0.f};

  const unsigned short* pa = A + (size_t)m * K;
  const unsigned short* pb[4];
#pragma unroll
  for (int t = 0; t < 4; ++t) pb[t] = Bt + (size_t)((tn0 + t) * 16 + l16) * K;

  for (int k0 = 0; k0 < K; k0 += 32) {
    __builtin_prefetch(pa + k0 + 64, 0, 1);           // global_prefetch_b8
    // Issue ALL loads for this k-step first, consume after: lets the
    // scheduler retire wmmas against partial loadcnt instead of draining.
    v8us a0 = *(const v8us*)(pa + k0 + hf * 8);
    v8us a1 = *(const v8us*)(pa + k0 + 16 + hf * 8);
    v16us braw[4];
#pragma unroll
    for (int t = 0; t < 4; ++t) braw[t] = __builtin_shufflevector(*(const v8us*)(pb[t] + k0 + hf * 8), *(const v8us*)(pb[t] + k0 + 16 + hf * 8), 0,1,2,3,4,5,6,7,8,9,10,11,12,13,14,15);

    v16bf av = __builtin_bit_cast(v16bf,
        __builtin_shufflevector(a0, a1, 0,1,2,3,4,5,6,7,8,9,10,11,12,13,14,15));
#pragma unroll
    for (int t = 0; t < 4; ++t) {
      acc[t] = __builtin_amdgcn_wmma_f32_16x16x32_bf16(
          false, av, false, __builtin_bit_cast(v16bf, braw[t]),
          (short)0, acc[t], false, false);
    }
  }

#pragma unroll
  for (int t = 0; t < 4; ++t) {
    int col = (tn0 + t) * 16 + l16;
    float bval = bias ? bias[col] : 0.0f;
    float* pc = C + (size_t)(tm * 16 + hf * 8) * Ncols + col;
#pragma unroll
    for (int r = 0; r < 8; ++r) pc[(size_t)r * Ncols] = acc[t][r] + bval;
  }
}

// Transpose-convert weight W[K x Nc] (f32) -> Wt[Nc x K] (bf16 bits).
__global__ void cvt_wt_bf16(const float* __restrict__ W, unsigned short* __restrict__ Wt,
                            int K, int Ncols)
{
  long i = (long)blockIdx.x * blockDim.x + threadIdx.x;
  if (i >= (long)K * Ncols) return;
  int n = (int)(i / K), k = (int)(i % K);
  Wt[i] = f2bf(W[(size_t)k * Ncols + n]);
}

// Row-major f32 -> bf16 conversion, 8 elements per thread (b128 in, b128 out).
__global__ void cvt_act_bf16(const float* __restrict__ X, unsigned short* __restrict__ Xb,
                             long n8)
{
  long i = (long)blockIdx.x * blockDim.x + threadIdx.x;
  if (i >= n8) return;
  const float4* p = (const float4*)(X + i * 8);
  float4 f0 = p[0], f1 = p[1];
  v8us o;
  o[0] = f2bf(f0.x); o[1] = f2bf(f0.y); o[2] = f2bf(f0.z); o[3] = f2bf(f0.w);
  o[4] = f2bf(f1.x); o[5] = f2bf(f1.y); o[6] = f2bf(f1.z); o[7] = f2bf(f1.w);
  *(v8us*)(Xb + i * 8) = o;
}

__global__ void fill_f32(float* __restrict__ p, long n, float v)
{
  long i = (long)blockIdx.x * blockDim.x + threadIdx.x;
  if (i < n) p[i] = v;
}

// asrc[n,h] = <h[n,h,:], a_src[h,:]> ; adst likewise.
__global__ void att_scores2(const float* __restrict__ h,
                            const float* __restrict__ aws, const float* __restrict__ awd,
                            float* __restrict__ os, float* __restrict__ od,
                            int Nn, int H, int Cper)
{
  int tid = blockIdx.x * blockDim.x + threadIdx.x;
  int n = tid / H, hh = tid % H;
  if (n >= Nn) return;
  const float* hp = h + (size_t)n * H * Cper + (size_t)hh * Cper;
  const float* ws = aws + hh * Cper;
  const float* wd = awd + hh * Cper;
  float s = 0.f, d = 0.f;
  for (int c = 0; c < Cper; ++c) { float v = hp[c]; s += v * ws[c]; d += v * wd[c]; }
  os[tid] = s; od[tid] = d;
}

// pass1: e = leaky_relu(asrc[src]+adst[dst]); segment max over dst.
__global__ void edge_pass1(const int* __restrict__ ei, int E, int Etot, int H,
                           const float* __restrict__ asrc, const float* __restrict__ adst,
                           float* __restrict__ evals, float* __restrict__ mx)
{
  long tid = (long)blockIdx.x * blockDim.x + threadIdx.x;
  int e = (int)(tid / H), h = (int)(tid % H);
  if (e >= Etot) return;
  int s, d;
  if (e < E) { s = ei[e]; d = ei[E + e]; } else { s = d = e - E; }
  float v = asrc[(size_t)s * H + h] + adst[(size_t)d * H + h];
  v = (v > 0.f) ? v : 0.2f * v;
  evals[tid] = v;
  atomicMaxF(&mx[(size_t)d * H + h], v);
}

// pass2: ex = exp(e - m[dst]); segment sum -> den; evals := ex.
__global__ void edge_pass2(const int* __restrict__ ei, int E, int Etot, int H,
                           float* __restrict__ evals, const float* __restrict__ mx,
                           float* __restrict__ den)
{
  long tid = (long)blockIdx.x * blockDim.x + threadIdx.x;
  int e = (int)(tid / H), h = (int)(tid % H);
  if (e >= Etot) return;
  int d = (e < E) ? ei[E + e] : (e - E);
  float ex = expf(evals[tid] - mx[(size_t)d * H + h]);
  evals[tid] = ex;
  atomicAdd(&den[(size_t)d * H + h], ex);
}

// out[dst] += h[src] * alpha ; one thread per (edge, 4 channels).
__global__ void edge_aggregate(const int* __restrict__ ei, int E, int Etot,
                               int H, int Cper,
                               const float* __restrict__ evals,
                               const float* __restrict__ den,
                               const float* __restrict__ hfeat,
                               float* __restrict__ out)
{
  const int F = H * Cper;
  const int groups = F >> 2;
  long tid = (long)blockIdx.x * blockDim.x + threadIdx.x;
  int e = (int)(tid / groups);
  int g = (int)(tid % groups);
  if (e >= Etot) return;
  int c = g * 4;
  int h = c / Cper;
  int s, d;
  if (e < E) { s = ei[e]; d = ei[E + e]; } else { s = d = e - E; }
  float alpha = evals[(size_t)e * H + h] / (den[(size_t)d * H + h] + 1e-16f);
  float4 hv = *(const float4*)(hfeat + (size_t)s * F + c);
  float* op = out + (size_t)d * F + c;
  atomicAdd(op + 0, hv.x * alpha);
  atomicAdd(op + 1, hv.y * alpha);
  atomicAdd(op + 2, hv.z * alpha);
  atomicAdd(op + 3, hv.w * alpha);
}

// Per-column mean and mean-of-squares (one block per column).
__global__ void bn_stats(const float* __restrict__ v, int Nrows, int Ncols,
                         float* __restrict__ mu, float* __restrict__ msq)
{
  __shared__ float s1[256], s2[256];
  int c = blockIdx.x;
  float a = 0.f, b = 0.f;
  for (int i = threadIdx.x; i < Nrows; i += 256) {
    float x = v[(size_t)i * Ncols + c];
    a += x; b += x * x;
  }
  s1[threadIdx.x] = a; s2[threadIdx.x] = b;
  __syncthreads();
  for (int st = 128; st > 0; st >>= 1) {
    if ((int)threadIdx.x < st) {
      s1[threadIdx.x] += s1[threadIdx.x + st];
      s2[threadIdx.x] += s2[threadIdx.x + st];
    }
    __syncthreads();
  }
  if (threadIdx.x == 0) { mu[c] = s1[0] / Nrows; msq[c] = s2[0] / Nrows; }
}

// v := relu(res + gamma*(v-mu)*rsqrt(var+eps) + beta)   (in-place over v)
__global__ void bn_res_relu(const float* __restrict__ res, float* __restrict__ v,
                            const float* __restrict__ mu, const float* __restrict__ msq,
                            const float* __restrict__ gam, const float* __restrict__ bet,
                            int Nrows, int Ncols)
{
  long tid = (long)blockIdx.x * blockDim.x + threadIdx.x;
  if (tid >= (long)Nrows * Ncols) return;
  int c = (int)(tid % Ncols);
  float m = mu[c];
  float var = msq[c] - m * m;
  float inv = rsqrtf(var + 1e-5f);
  float y = res[tid] + (v[tid] - m) * inv * gam[c] + bet[c];
  v[tid] = (y > 0.f) ? y : 0.f;
}

// out[t,:7] = (h7[src]*h7[dst]) @ fc4_w + fc4_b ; one wave per train edge.
__global__ void pair_fc4(const int* __restrict__ ei, int E,
                         const int* __restrict__ teid, int TE,
                         const float* __restrict__ h7,
                         const float* __restrict__ w, const float* __restrict__ bias,
                         float* __restrict__ out)
{
  int gtid = blockIdx.x * blockDim.x + threadIdx.x;
  int t    = gtid >> 5;
  int lane = gtid & 31;
  if (t >= TE) return;
  int eid = teid[t];
  int s = ei[eid];
  int d = ei[E + eid];
  const float* hs = h7 + (size_t)s * 256;
  const float* hd = h7 + (size_t)d * 256;
  float acc[7] = {0.f, 0.f, 0.f, 0.f, 0.f, 0.f, 0.f};
  for (int k = lane; k < 256; k += 32) {
    float p = hs[k] * hd[k];
    const float* wr = w + k * 7;
#pragma unroll
    for (int j = 0; j < 7; ++j) acc[j] += p * wr[j];
  }
#pragma unroll
  for (int j = 0; j < 7; ++j) {
#pragma unroll
    for (int o = 16; o > 0; o >>= 1) acc[j] += __shfl_down(acc[j], o);
  }
  if (lane == 0) {
#pragma unroll
    for (int j = 0; j < 7; ++j) out[(size_t)t * 7 + j] = acc[j] + bias[j];
  }
}

static inline unsigned gblk(long n) { return (unsigned)((n + 255) / 256); }

extern "C" void kernel_launch(void* const* d_in, const int* in_sizes, int n_in,
                              void* d_out, int out_size, void* d_ws, size_t ws_size,
                              hipStream_t stream)
{
  const int N = 10000, E = 160000, TE = 65536, ETOT = E + N;
  const int F1 = 384, F2 = 256, H1 = 8, C1 = 48;
  (void)in_sizes; (void)n_in; (void)out_size; (void)ws_size;

  const float* x      = (const float*)d_in[0];
  const int*   ei     = (const int*)  d_in[1];
  const int*   teid   = (const int*)  d_in[2];
  const float* fc1_w  = (const float*)d_in[3];
  const float* fc1_b  = (const float*)d_in[4];
  const float* fc5_w  = (const float*)d_in[5];
  const float* fc5_b  = (const float*)d_in[6];
  const float* fc2_w  = (const float*)d_in[7];
  const float* fc2_b  = (const float*)d_in[8];
  const float* fc4_w  = (const float*)d_in[9];
  const float* fc4_b  = (const float*)d_in[10];
  const float* gat1_w = (const float*)d_in[11];
  const float* g1_as  = (const float*)d_in[12];
  const float* g1_ad  = (const float*)d_in[13];
  // d_in[14] = gat1_b : cancels in batchnorm
  const float* gat2_w = (const float*)d_in[15];
  const float* g2_as  = (const float*)d_in[16];
  const float* g2_ad  = (const float*)d_in[17];
  // d_in[18] = gat2_b : cancels in batchnorm
  const float* bn1_g  = (const float*)d_in[19];
  const float* bn1_b  = (const float*)d_in[20];
  const float* bn2_g  = (const float*)d_in[21];
  const float* bn2_b  = (const float*)d_in[22];

  char* ws = (char*)d_ws;
  size_t off = 0;
  auto alloc = [&](size_t bytes) -> char* {
    char* p = ws + off;
    off = (off + bytes + 255) & ~(size_t)255;
    return p;
  };
  float* bufA = (float*)alloc((size_t)N * F1 * 4);   // xg1 -> gout1 -> hmid
  float* bufB = (float*)alloc((size_t)N * F1 * 4);   // h1  -> h5
  float* bufC = (float*)alloc((size_t)N * F2 * 4);   // h2  -> h7
  float* bufD = (float*)alloc((size_t)N * F2 * 4);   // gout2 -> h6
  float* ev1  = (float*)alloc((size_t)ETOT * H1 * 4);
  float* ev2  = (float*)alloc((size_t)ETOT * 4);
  float* as1  = (float*)alloc((size_t)N * H1 * 4);
  float* ad1  = (float*)alloc((size_t)N * H1 * 4);
  float* mx1  = (float*)alloc((size_t)N * H1 * 4);
  float* dn1  = (float*)alloc((size_t)N * H1 * 4);
  float* as2  = (float*)alloc((size_t)N * 4);
  float* ad2  = (float*)alloc((size_t)N * 4);
  float* mx2  = (float*)alloc((size_t)N * 4);
  float* dn2  = (float*)alloc((size_t)N * 4);
  float* mu   = (float*)alloc((size_t)F1 * 4);
  float* msq  = (float*)alloc((size_t)F1 * 4);
  unsigned short* actbf  = (unsigned short*)alloc((size_t)N * F1 * 2);  // bf16 A scratch
  unsigned short* wt_fc1 = (unsigned short*)alloc((size_t)F1 * F1 * 2);
  unsigned short* wt_g1  = (unsigned short*)alloc((size_t)F1 * F1 * 2);
  unsigned short* wt_fc5 = (unsigned short*)alloc((size_t)F1 * F2 * 2);
  unsigned short* wt_g2  = (unsigned short*)alloc((size_t)F2 * F2 * 2);
  unsigned short* wt_fc2 = (unsigned short*)alloc((size_t)F2 * F2 * 2);

  auto cvt = [&](const float* W, unsigned short* Wt, int K, int Nc) {
    cvt_wt_bf16<<<gblk((long)K * Nc), 256, 0, stream>>>(W, Wt, K, Nc);
  };
  cvt(fc1_w,  wt_fc1, F1, F1);
  cvt(gat1_w, wt_g1,  F1, F1);
  cvt(fc5_w,  wt_fc5, F1, F2);
  cvt(gat2_w, wt_g2,  F2, F2);
  cvt(fc2_w,  wt_fc2, F2, F2);

  // cvt activation (M x K f32 -> bf16), then WMMA GEMM.
  auto gemm = [&](const float* A, const unsigned short* Bt, const float* bias,
                  float* C, int M, int K, int Nc) {
    long n8 = (long)M * K / 8;
    cvt_act_bf16<<<gblk(n8), 256, 0, stream>>>(A, actbf, n8);
    dim3 grid(Nc / 128, M / 16);
    gemm_wmma_bf16<<<grid, 64, 0, stream>>>(actbf, Bt, bias, C, M, K, Nc);
  };

  // ---- Layer 1: fc1 -> GAT(8 heads x 48) -> BN -> relu(x + .) ----
  gemm(x,    wt_fc1, fc1_b,  bufA, N, F1, F1);                 // xg1
  gemm(bufA, wt_g1,  nullptr, bufB, N, F1, F1);                // h1
  att_scores2<<<gblk((long)N * H1), 256, 0, stream>>>(bufB, g1_as, g1_ad, as1, ad1, N, H1, C1);
  fill_f32<<<gblk((long)N * H1), 256, 0, stream>>>(mx1, (long)N * H1, -FLT_MAX);
  fill_f32<<<gblk((long)N * H1), 256, 0, stream>>>(dn1, (long)N * H1, 0.f);
  fill_f32<<<gblk((long)N * F1), 256, 0, stream>>>(bufA, (long)N * F1, 0.f);  // gout1
  edge_pass1<<<gblk((long)ETOT * H1), 256, 0, stream>>>(ei, E, ETOT, H1, as1, ad1, ev1, mx1);
  edge_pass2<<<gblk((long)ETOT * H1), 256, 0, stream>>>(ei, E, ETOT, H1, ev1, mx1, dn1);
  edge_aggregate<<<gblk((long)ETOT * (F1 / 4)), 256, 0, stream>>>(ei, E, ETOT, H1, C1, ev1, dn1, bufB, bufA);
  bn_stats<<<F1, 256, 0, stream>>>(bufA, N, F1, mu, msq);
  bn_res_relu<<<gblk((long)N * F1), 256, 0, stream>>>(x, bufA, mu, msq, bn1_g, bn1_b, N, F1); // hmid

  // ---- Layer 2: fc5 -> GAT(1 head x 256) -> BN -> relu(h5 + .) ----
  gemm(bufA, wt_fc5, fc5_b,  bufB, N, F1, F2);                 // h5
  gemm(bufB, wt_g2,  nullptr, bufC, N, F2, F2);                // h2
  att_scores2<<<gblk((long)N), 256, 0, stream>>>(bufC, g2_as, g2_ad, as2, ad2, N, 1, F2);
  fill_f32<<<gblk((long)N), 256, 0, stream>>>(mx2, (long)N, -FLT_MAX);
  fill_f32<<<gblk((long)N), 256, 0, stream>>>(dn2, (long)N, 0.f);
  fill_f32<<<gblk((long)N * F2), 256, 0, stream>>>(bufD, (long)N * F2, 0.f);  // gout2
  edge_pass1<<<gblk((long)ETOT), 256, 0, stream>>>(ei, E, ETOT, 1, as2, ad2, ev2, mx2);
  edge_pass2<<<gblk((long)ETOT), 256, 0, stream>>>(ei, E, ETOT, 1, ev2, mx2, dn2);
  edge_aggregate<<<gblk((long)ETOT * (F2 / 4)), 256, 0, stream>>>(ei, E, ETOT, 1, F2, ev2, dn2, bufC, bufD);
  bn_stats<<<F2, 256, 0, stream>>>(bufD, N, F2, mu, msq);
  bn_res_relu<<<gblk((long)N * F2), 256, 0, stream>>>(bufB, bufD, mu, msq, bn2_g, bn2_b, N, F2); // h6

  // ---- Head: fc2 -> pairwise product -> fc4 ----
  gemm(bufD, wt_fc2, fc2_b, bufC, N, F2, F2);                  // h7
  pair_fc4<<<gblk((long)TE * 32), 256, 0, stream>>>(ei, E, teid, TE, bufC, fc4_w, fc4_b, (float*)d_out);
}
